// MultiHeadAttention_90675349553877
// MI455X (gfx1250) — matrix operand order repaired and checked by reading, not compiled, not run on hardware
//
#include <hip/hip_runtime.h>

#define DEV __device__ __forceinline__

typedef __attribute__((ext_vector_type(16))) __bf16 v16bf;
typedef __attribute__((ext_vector_type(8)))  float  v8f;
typedef __attribute__((ext_vector_type(4)))  unsigned int u32x4;
typedef __attribute__((ext_vector_type(8)))  int i32x8;
typedef __attribute__((ext_vector_type(4)))  int i32x4;

static constexpr int   Bb = 2, Nn = 2048, Dd = 1024, Hh = 16, DHd = 64;
static constexpr float SCALE = 0.125f;   // 1/sqrt(64)
static constexpr float MASKV = -1.0e7f;
static constexpr float NEGBIG = -3.0e38f;

#if defined(__has_builtin)
#if __has_builtin(__builtin_amdgcn_tensor_load_to_lds) && __has_builtin(__builtin_amdgcn_s_wait_tensorcnt)
#define HAVE_TDM 1
#endif
#endif
#ifndef HAVE_TDM
#define HAVE_TDM 0
#endif

union BF16x16 { v16bf v; unsigned int u[8]; };

DEV unsigned int f32_bf16(float f) {
  unsigned int u = __float_as_uint(f);
  u += 0x7FFFu + ((u >> 16) & 1u);           // round-to-nearest-even
  return u >> 16;
}
DEV unsigned int pack_bf16(float lo, float hi) {
  return f32_bf16(lo) | (f32_bf16(hi) << 16);
}

// CDNA5 WMMA 16-bit fragment k-offsets (CDNA5 ISA 7.12.2), per VGPR j, lane-half
DEV int a_koff(int j, int half) { return (j < 4 ? 2 * j : 2 * j + 8) + 8 * half; }
DEV int b_koff(int j, int half) { return (j < 4 ? 2 * j : 2 * j + 8) + 8 * half; }

#if HAVE_TDM
// 2D tile DMA global->LDS via Tensor Data Mover.  data_size = 2 bytes (bf16).
// width/height/rstride in elements; tile assumed fully in-bounds.
DEV void tdm_load_2d(const void* gptr, unsigned lds_byte_off, int width, int height,
                     long long rstride) {
  unsigned long long ga = (unsigned long long)gptr;
  u32x4 g0;
  g0[0] = 1u;                                          // count=1, user descriptor
  g0[1] = lds_byte_off;                                // lds_addr (bytes)
  g0[2] = (unsigned)(ga & 0xffffffffu);                // global_addr[31:0]
  g0[3] = (unsigned)((ga >> 32) & 0x1ffffffu) | (2u << 30);  // addr[56:32] | type=2
  i32x8 g1;
  g1[0] = 1 << 16;                                     // data_size=1 (2 bytes)
  g1[1] = (width & 0xffff) << 16;                      // tensor_dim0[15:0]
  g1[2] = ((width >> 16) & 0xffff) | ((height & 0xffff) << 16);   // dim0 hi | dim1 lo
  g1[3] = ((height >> 16) & 0xffff) | ((width & 0xffff) << 16);   // dim1 hi | tile_dim0
  g1[4] = height & 0xffff;                             // tile_dim1 (tile_dim2=0)
  g1[5] = (int)(rstride & 0xffffffffll);               // tensor_dim0_stride[31:0]
  g1[6] = (int)((rstride >> 32) & 0xffffll);           // stride[47:32]; dim1_stride=0
  g1[7] = 0;
  i32x4 z4 = {0, 0, 0, 0};
  i32x8 z8 = {0, 0, 0, 0, 0, 0, 0, 0};
  __builtin_amdgcn_tensor_load_to_lds(g0, g1, z4, z4, z8, 0);
}
template <typename T>
DEV unsigned lds_byteoff(T* p) { return (unsigned)(unsigned long long)(void*)p; }
#endif

// ---------------------------------------------------------------------------
// Projection GEMM: (X f32 4096x1024) x (W f32 1024x1024) -> bf16, head-split.
// MODE 0: Y[bh][n][dh] (Q,K)    MODE 1: Y[bh][dh][n] (V, pre-transposed)
// 256 threads = 8 waves; WG tile 128x128; K-step 32; wmma f32_16x16x32_bf16
// ---------------------------------------------------------------------------
template <int MODE>
__global__ __launch_bounds__(256) void proj_qkv(const float* __restrict__ X,
                                                const float* __restrict__ W,
                                                unsigned short* __restrict__ Y) {
  __shared__ __align__(16) unsigned short As[128][32];   // [row][k]
  __shared__ __align__(16) unsigned short Bs[128][32];   // [col][k] (pre-transposed)
  const int tid = threadIdx.x, wave = tid >> 5, lane = tid & 31;
  const int half = lane >> 4, m = lane & 15;
  const int row0 = blockIdx.y * 128, col0 = blockIdx.x * 128;

  v8f acc[8];
#pragma unroll
  for (int t = 0; t < 8; ++t)
#pragma unroll
    for (int r = 0; r < 8; ++r) acc[t][r] = 0.0f;

  for (int kt = 0; kt < Dd; kt += 32) {
    __syncthreads();
#pragma unroll
    for (int i = 0; i < 8; ++i) {        // 2048 bf16-pairs each for A and B
      int e = i * 256 + tid;
      { int r = e >> 4, c = (e & 15) * 2;
        const float* xp = &X[(size_t)(row0 + r) * Dd + kt + c];
        *(unsigned int*)&As[r][c] = pack_bf16(xp[0], xp[1]); }
      { int cc = e & 127, k = (e >> 7) * 2;
        const float* wp = &W[(size_t)(kt + k) * Dd + col0 + cc];
        *(unsigned int*)&Bs[cc][k] = pack_bf16(wp[0], wp[Dd]); }
    }
    __syncthreads();

    BF16x16 a;
#pragma unroll
    for (int j = 0; j < 8; ++j)
      a.u[j] = *(const unsigned int*)&As[16 * wave + m][a_koff(j, half)];
#pragma unroll
    for (int t = 0; t < 8; ++t) {
      BF16x16 bm;
#pragma unroll
      for (int j = 0; j < 8; ++j)
        bm.u[j] = *(const unsigned int*)&Bs[16 * t + m][b_koff(j, half)];
      acc[t] = __builtin_amdgcn_wmma_f32_16x16x32_bf16(false, a.v, false, bm.v,
                                                       (short)0, acc[t], false, false);
    }
  }

#pragma unroll
  for (int t = 0; t < 8; ++t)
#pragma unroll
    for (int r = 0; r < 8; ++r) {
      int row = row0 + 16 * wave + r + 8 * half;       // b*N + n
      int col = col0 + 16 * t + m;                     // h*64 + dh
      int bb = row >> 11, nq = row & (Nn - 1);
      int hh = col >> 6,  dh = col & (DHd - 1);
      size_t idx = (MODE == 0)
          ? (((size_t)(bb * Hh + hh)) * Nn + nq) * DHd + dh
          : (((size_t)(bb * Hh + hh)) * DHd + dh) * Nn + nq;
      Y[idx] = (unsigned short)f32_bf16(acc[t][r]);
    }
}

// ---------------------------------------------------------------------------
// Output projection: out_f32[4096][1024] = (heads bf16 4096x1024) x (Wo f32)
// ---------------------------------------------------------------------------
__global__ __launch_bounds__(256) void out_proj(const unsigned short* __restrict__ Hd,
                                                const float* __restrict__ W,
                                                float* __restrict__ out) {
  __shared__ __align__(16) unsigned short As[128][32];
  __shared__ __align__(16) unsigned short Bs[128][32];
  const int tid = threadIdx.x, wave = tid >> 5, lane = tid & 31;
  const int half = lane >> 4, m = lane & 15;
  const int row0 = blockIdx.y * 128, col0 = blockIdx.x * 128;

  v8f acc[8];
#pragma unroll
  for (int t = 0; t < 8; ++t)
#pragma unroll
    for (int r = 0; r < 8; ++r) acc[t][r] = 0.0f;

  for (int kt = 0; kt < Dd; kt += 32) {
    __syncthreads();
#pragma unroll
    for (int i = 0; i < 8; ++i) {
      int e = i * 256 + tid;
      { int r = e >> 4, c = (e & 15) * 2;        // heads already bf16: u32 copy
        *(unsigned int*)&As[r][c] =
            *(const unsigned int*)&Hd[(size_t)(row0 + r) * Dd + kt + c]; }
      { int cc = e & 127, k = (e >> 7) * 2;
        const float* wp = &W[(size_t)(kt + k) * Dd + col0 + cc];
        *(unsigned int*)&Bs[cc][k] = pack_bf16(wp[0], wp[Dd]); }
    }
    __syncthreads();

    BF16x16 a;
#pragma unroll
    for (int j = 0; j < 8; ++j)
      a.u[j] = *(const unsigned int*)&As[16 * wave + m][a_koff(j, half)];
#pragma unroll
    for (int t = 0; t < 8; ++t) {
      BF16x16 bm;
#pragma unroll
      for (int j = 0; j < 8; ++j)
        bm.u[j] = *(const unsigned int*)&Bs[16 * t + m][b_koff(j, half)];
      acc[t] = __builtin_amdgcn_wmma_f32_16x16x32_bf16(false, a.v, false, bm.v,
                                                       (short)0, acc[t], false, false);
    }
  }

#pragma unroll
  for (int t = 0; t < 8; ++t)
#pragma unroll
    for (int r = 0; r < 8; ++r) {
      int row = row0 + 16 * wave + r + 8 * half;
      int col = col0 + 16 * t + m;
      out[(size_t)row * Dd + col] = acc[t][r];
    }
}

// ---------------------------------------------------------------------------
// Flash attention over one (b,h): wave owns 16 queries, streams 32-key blocks.
// K/V tiles DMA'd into LDS by the Tensor Data Mover (wave 0), completion via
// s_wait_tensorcnt.  grid = (N/128, B*H); block = 256 (8 waves)
// ---------------------------------------------------------------------------
__global__ __launch_bounds__(256) void attn(const unsigned short* __restrict__ Qb,
                                            const unsigned short* __restrict__ Kb,
                                            const unsigned short* __restrict__ Vt,
                                            const int* __restrict__ vlen,
                                            unsigned short* __restrict__ Hd) {
  __shared__ __align__(16) unsigned short Ks[32][64];     // [key][dh]
  __shared__ __align__(16) unsigned short Vs[64][32];     // [dh][key]
  __shared__ __align__(16) unsigned short Ps[8][16][32];  // per-wave P staging [q][key]
  const int tid = threadIdx.x, wave = tid >> 5, lane = tid & 31;
  const int half = lane >> 4, m = lane & 15;
  const int bh = blockIdx.y;
  const int vl = vlen[bh];
  const unsigned short* Qp = Qb + (size_t)bh * Nn * DHd;
  const unsigned short* Kp = Kb + (size_t)bh * Nn * DHd;
  const unsigned short* Vp = Vt + (size_t)bh * Nn * DHd;  // [dh][n]
  const int q0 = blockIdx.x * 128 + wave * 16;

  // Q fragments: A-matrix 16x64 bf16 = two 16x32 frags, direct from global
  BF16x16 qf[2];
#pragma unroll
  for (int kk = 0; kk < 2; ++kk)
#pragma unroll
    for (int j = 0; j < 8; ++j)
      qf[kk].u[j] = *(const unsigned int*)&Qp[(size_t)(q0 + m) * DHd + 32 * kk + a_koff(j, half)];

  v8f o[4];
  float rmax[8], rsum[8];
#pragma unroll
  for (int t = 0; t < 4; ++t)
#pragma unroll
    for (int r = 0; r < 8; ++r) o[t][r] = 0.0f;
#pragma unroll
  for (int r = 0; r < 8; ++r) { rmax[r] = NEGBIG; rsum[r] = 0.0f; }

  for (int k0 = 0; k0 < vl; k0 += 32) {
    __syncthreads();                       // previous block's LDS reads done
#if HAVE_TDM
    if (wave == 0) {
      tdm_load_2d(Kp + (size_t)k0 * DHd, lds_byteoff(&Ks[0][0]), DHd, 32, DHd);
      tdm_load_2d(Vp + k0,               lds_byteoff(&Vs[0][0]), 32, DHd, Nn);
      __builtin_amdgcn_s_wait_tensorcnt(0);
    }
#else
#pragma unroll
    for (int i = 0; i < 4; ++i) {          // cooperative u32 copies (fallback)
      int e = i * 256 + tid;               // 1024 u32 per tile
      ((unsigned int*)Ks)[e] =
          ((const unsigned int*)(Kp + (size_t)k0 * DHd))[e];
      { int r = e >> 4, c = (e & 15) * 2;
        *(unsigned int*)&Vs[r][c] = *(const unsigned int*)&Vp[(size_t)r * Nn + k0 + c]; }
    }
#endif
    __syncthreads();

    // S = Q * K^T : two 16x16 score tiles, contraction dh=64 (2 x k=32)
    v8f s[2];
    float bmax[8];
#pragma unroll
    for (int r = 0; r < 8; ++r) bmax[r] = NEGBIG;
#pragma unroll
    for (int t = 0; t < 2; ++t) {
      v8f st;
#pragma unroll
      for (int r = 0; r < 8; ++r) st[r] = 0.0f;
#pragma unroll
      for (int kk = 0; kk < 2; ++kk) {
        BF16x16 bk;
#pragma unroll
        for (int j = 0; j < 8; ++j)
          bk.u[j] = *(const unsigned int*)&Ks[16 * t + m][32 * kk + b_koff(j, half)];
        st = __builtin_amdgcn_wmma_f32_16x16x32_bf16(false, qf[kk].v, false, bk.v,
                                                     (short)0, st, false, false);
      }
      int keycol = k0 + 16 * t + m;
#pragma unroll
      for (int r = 0; r < 8; ++r) {
        float sv = st[r] * SCALE;
        if (keycol >= vl) sv = MASKV;
        st[r] = sv;
        bmax[r] = fmaxf(bmax[r], sv);
      }
      s[t] = st;
    }

    // row-max across the 16 lanes holding each row (butterfly stays in half)
#pragma unroll
    for (int r = 0; r < 8; ++r) {
      float v = bmax[r];
      v = fmaxf(v, __shfl_xor(v, 1, 32));
      v = fmaxf(v, __shfl_xor(v, 2, 32));
      v = fmaxf(v, __shfl_xor(v, 4, 32));
      v = fmaxf(v, __shfl_xor(v, 8, 32));
      bmax[r] = v;
    }
    float alpha[8];
#pragma unroll
    for (int r = 0; r < 8; ++r) {
      float mn = fmaxf(rmax[r], bmax[r]);
      alpha[r] = __expf(rmax[r] - mn);
      rmax[r] = mn;
    }
    float psum[8];
#pragma unroll
    for (int r = 0; r < 8; ++r) {
      float p0 = __expf(s[0][r] - rmax[r]);
      float p1 = __expf(s[1][r] - rmax[r]);
      Ps[wave][r + 8 * half][m]      = (unsigned short)f32_bf16(p0);
      Ps[wave][r + 8 * half][16 + m] = (unsigned short)f32_bf16(p1);
      psum[r] = p0 + p1;
    }
#pragma unroll
    for (int r = 0; r < 8; ++r) {
      float v = psum[r];
      v += __shfl_xor(v, 1, 32);
      v += __shfl_xor(v, 2, 32);
      v += __shfl_xor(v, 4, 32);
      v += __shfl_xor(v, 8, 32);
      rsum[r] = rsum[r] * alpha[r] + v;
    }
#pragma unroll
    for (int t = 0; t < 4; ++t)
#pragma unroll
      for (int r = 0; r < 8; ++r) o[t][r] *= alpha[r];

    __syncthreads();   // Ps C-layout -> A-layout round trip through LDS

    BF16x16 pf;
#pragma unroll
    for (int j = 0; j < 8; ++j)
      pf.u[j] = *(const unsigned int*)&Ps[wave][m][a_koff(j, half)];
#pragma unroll
    for (int t = 0; t < 4; ++t) {
      BF16x16 bv;
#pragma unroll
      for (int j = 0; j < 8; ++j)
        bv.u[j] = *(const unsigned int*)&Vs[16 * t + m][b_koff(j, half)];
      o[t] = __builtin_amdgcn_wmma_f32_16x16x32_bf16(false, pf.v, false, bv.v,
                                                     (short)0, o[t], false, false);
    }
  }

  // epilogue: heads[b][n][h*64+dh] = O / rowsum  (bf16 for the output GEMM)
  const int b = bh >> 4, hh = bh & 15;
#pragma unroll
  for (int t = 0; t < 4; ++t)
#pragma unroll
    for (int r = 0; r < 8; ++r) {
      int q = q0 + r + 8 * half;
      int dcol = hh * DHd + 16 * t + m;
      Hd[((size_t)b * Nn + q) * Dd + dcol] = (unsigned short)f32_bf16(o[t][r] / rsum[r]);
    }
}

// ---------------------------------------------------------------------------
extern "C" void kernel_launch(void* const* d_in, const int* in_sizes, int n_in,
                              void* d_out, int out_size, void* d_ws, size_t ws_size,
                              hipStream_t stream) {
  (void)in_sizes; (void)n_in; (void)out_size; (void)ws_size;
  const float* q  = (const float*)d_in[0];
  const float* k  = (const float*)d_in[1];
  const float* v  = (const float*)d_in[2];
  const int*   vl = (const int*)d_in[3];
  const float* Wq = (const float*)d_in[4];
  const float* Wk = (const float*)d_in[5];
  const float* Wv = (const float*)d_in[6];
  const float* Wo = (const float*)d_in[7];
  float* out = (float*)d_out;

  const size_t per = (size_t)Bb * Hh * Nn * DHd;   // bf16 elements per tensor
  unsigned short* Qb = (unsigned short*)d_ws;
  unsigned short* Kb = Qb + per;
  unsigned short* Vb = Kb + per;     // V stored transposed per head: [bh][dh][n]
  unsigned short* Hd = Vb + per;

  dim3 blk(256);
  dim3 gproj(Dd / 128, (Bb * Nn) / 128);     // (8, 32)
  dim3 gattn(Nn / 128, Bb * Hh);             // (16, 32)

  proj_qkv<0><<<gproj, blk, 0, stream>>>(q, Wq, Qb);
  proj_qkv<0><<<gproj, blk, 0, stream>>>(k, Wk, Kb);
  proj_qkv<1><<<gproj, blk, 0, stream>>>(v, Wv, Vb);
  attn<<<gattn, blk, 0, stream>>>(Qb, Kb, Vb, vl, Hd);
  out_proj<<<gproj, blk, 0, stream>>>(Hd, Wo, out);
}
